// SelfAttention_10033043604013
// MI455X (gfx1250) — matrix operand order repaired and checked by reading, not compiled, not run on hardware
//
#include <hip/hip_runtime.h>
#include <hip/hip_bf16.h>

typedef __attribute__((ext_vector_type(16))) __bf16 v16bf;
typedef __attribute__((ext_vector_type(8)))  __bf16 v8bf;
typedef __attribute__((ext_vector_type(8)))  float  v8f;

#define B_    2
#define T_    2048
#define C_    4096
#define H_    32
#define DH_   128
#define NKV_  8
#define NQKV_ 6144                 // 4096 (q) + 1024 (k) + 1024 (v)
#define SCALE_ 0.08838834764831845f  // 1/sqrt(128)

// ---------------------------------------------------------------------------
// WMMA helpers (CDNA5 wave32, 16x16x32 bf16 -> f32)
// ---------------------------------------------------------------------------
__device__ __forceinline__ v8f wmma_bf16(v16bf a, v16bf b, v8f c) {
  // args: (neg_a, A, neg_b, B, c_mod, C, reuse_a, reuse_b)
  return __builtin_amdgcn_wmma_f32_16x16x32_bf16(false, a, false, b,
                                                 (short)0, c, false, false);
}

// A-fragment, 16x32 bf16, row-major source with leading dim `ld` (elements).
// Lanes 0-15: row=lane, K = 0..15.  Lanes 16-31: row=lane-16,
// K = 16..31 (contiguous, matches load_b_frag).
__device__ __forceinline__ v16bf load_a_frag(const __bf16* base, int ld) {
  const int lane = threadIdx.x & 31;
  const int row  = lane & 15;
  const int hi   = lane >> 4;
  const __bf16* p = base + row * ld + hi * 16;
  union { v16bf v; v8bf h[2]; } u;
  u.h[0] = *(const v8bf*)(p);        // K + 0..7
  u.h[1] = *(const v8bf*)(p + 8);    // K + 8..15
  return u.v;
}

// B-fragment, 32x16 bf16, from an N-major store: element (n,k) at
// base[n*ld + k].  Lane holds column N=lane%16, K = 16*(lane/16) + 0..15
// consecutive  ->  one contiguous 32-byte read per lane.
__device__ __forceinline__ v16bf load_b_frag(const __bf16* base, int ld) {
  const int lane = threadIdx.x & 31;
  const int col  = lane & 15;
  const int hi   = lane >> 4;
  return *(const v16bf*)(base + (size_t)col * ld + hi * 16);
}

// ---------------------------------------------------------------------------
// CDNA5 async global -> LDS (ASYNCcnt-tracked, bypasses VGPRs).
// VDST VGPR holds the wave-relative LDS byte offset = low 32 bits of the
// generic shared-memory address (ISA 10.2).
// ---------------------------------------------------------------------------
__device__ __forceinline__ void async_ld_b128(const __bf16* lds_generic,
                                              const __bf16* g) {
  uint32_t l = (uint32_t)(size_t)lds_generic;
  asm volatile("global_load_async_to_lds_b128 %0, %1, off"
               :: "v"(l), "v"(g) : "memory");
}
__device__ __forceinline__ void wait_async0() {
  asm volatile("s_wait_asynccnt 0" ::: "memory");
}

// Cooperative async prefetch of a 32x128 bf16 tile (8 KB), 256 threads,
// two b128 transfers per thread.
__device__ __forceinline__ void prefetch_tile(__bf16* lds_buf,
                                              const __bf16* g_base,
                                              size_t gld) {
#pragma unroll
  for (int i = 0; i < 2; ++i) {
    const int idx = threadIdx.x + i * 256;
    const int rr  = idx >> 4;          // 0..31
    const int cc  = (idx & 15) * 8;    // 0..120
    async_ld_b128(lds_buf + rr * 128 + cc, g_base + (size_t)rr * gld + cc);
  }
}

// ---------------------------------------------------------------------------
// Kernel 0: fp32 -> bf16 convert of x
// ---------------------------------------------------------------------------
__global__ void cvt_x_kernel(const float* __restrict__ x,
                             __bf16* __restrict__ xb, size_t n) {
  size_t i      = (size_t)blockIdx.x * blockDim.x + threadIdx.x;
  size_t stride = (size_t)gridDim.x * blockDim.x;
  for (; i < n; i += stride) xb[i] = (__bf16)x[i];
}

// ---------------------------------------------------------------------------
// Kernel 1: pack weights to bf16, N-major (transposed):
//   wqkvT[n][k], n in [0,6144): q cols, then k cols, then v cols
//   woT[n][k]
// ---------------------------------------------------------------------------
__global__ void pack_w_kernel(const float* __restrict__ wq,
                              const float* __restrict__ wk,
                              const float* __restrict__ wv,
                              const float* __restrict__ wo,
                              __bf16* __restrict__ wqkvT,
                              __bf16* __restrict__ woT) {
  const size_t NQK = (size_t)NQKV_ * C_;        // 25,165,824
  const size_t NWO = (size_t)C_ * C_;           // 16,777,216
  size_t i      = (size_t)blockIdx.x * blockDim.x + threadIdx.x;
  size_t stride = (size_t)gridDim.x * blockDim.x;
  for (; i < NQK + NWO; i += stride) {
    if (i < NQK) {
      int n = (int)(i / C_);
      int k = (int)(i % C_);
      float v;
      if (n < 4096)      v = wq[(size_t)k * 4096 + n];
      else if (n < 5120) v = wk[(size_t)k * 1024 + (n - 4096)];
      else               v = wv[(size_t)k * 1024 + (n - 5120)];
      wqkvT[i] = (__bf16)v;
    } else {
      size_t j = i - NQK;
      int n = (int)(j / C_);
      int k = (int)(j % C_);
      woT[j] = (__bf16)wo[(size_t)k * C_ + n];
    }
  }
}

// ---------------------------------------------------------------------------
// RoPE / store epilogue for one 16x16 C tile of the fused QKV output.
// All branch conditions are wave-uniform (16-col tiles never straddle the
// 128-aligned q/k/v region boundaries).
// ---------------------------------------------------------------------------
__device__ __forceinline__ void qkv_epilogue(
    v8f acc, int n, int bb, int tbase, int hi,
    const float* __restrict__ cosT, const float* __restrict__ sinT,
    __bf16* __restrict__ qO, __bf16* __restrict__ kO,
    __bf16* __restrict__ vTO) {
  if (n < 4096) {                          // ---- Q: RoPE + 1/sqrt(Dh)
    const int h = n >> 7, d = n & 127;
#pragma unroll
    for (int r = 0; r < 8; ++r) {
      const int t = tbase + r + 8 * hi;
      float v = acc[r];
      float o = __shfl_xor(v, 1, 32);      // partner column of the RoPE pair
      float c = cosT[t * DH_ + d];
      float s = sinT[t * DH_ + d];
      float res = (n & 1) ? (o * s + v * c) : (v * c - o * s);
      qO[((size_t)(bb * H_ + h) * T_ + t) * DH_ + d] = (__bf16)(res * SCALE_);
    }
  } else if (n < 5120) {                   // ---- K: RoPE
    const int kh = (n - 4096) >> 7, d = n & 127;
#pragma unroll
    for (int r = 0; r < 8; ++r) {
      const int t = tbase + r + 8 * hi;
      float v = acc[r];
      float o = __shfl_xor(v, 1, 32);
      float c = cosT[t * DH_ + d];
      float s = sinT[t * DH_ + d];
      float res = (n & 1) ? (o * s + v * c) : (v * c - o * s);
      kO[((size_t)(bb * NKV_ + kh) * T_ + t) * DH_ + d] = (__bf16)res;
    }
  } else {                                 // ---- V: transposed store [Dh][T]
    const int kh = (n - 5120) >> 7, d = n & 127;
#pragma unroll
    for (int r = 0; r < 8; ++r) {
      const int t = tbase + r + 8 * hi;
      vTO[((size_t)(bb * NKV_ + kh) * DH_ + d) * T_ + t] = (__bf16)acc[r];
    }
  }
}

// ---------------------------------------------------------------------------
// Kernel 2: fused QKV projection (bf16 WMMA) + RoPE.
// Block = 256 threads = 8 waves; tile = 32 rows x 128 cols.  Each wave keeps
// two accumulators (M=0..15 and M=16..31) and reuses every weight B-fragment
// twice.  x tile double-buffered in LDS via async global->LDS loads.
// ---------------------------------------------------------------------------
__global__ void qkv_rope_kernel(const __bf16* __restrict__ xb,
                                const __bf16* __restrict__ wT,
                                const float* __restrict__ cosT,
                                const float* __restrict__ sinT,
                                __bf16* __restrict__ qO,
                                __bf16* __restrict__ kO,
                                __bf16* __restrict__ vTO) {
  __shared__ alignas(32) __bf16 sx[2][32 * 128];   // 2 x 8 KB

  const int wave    = threadIdx.x >> 5;
  const int lane    = threadIdx.x & 31;
  const int col     = lane & 15;
  const int hi      = lane >> 4;
  const int n0      = blockIdx.x * 128 + wave * 16;   // fused output column
  const int rowBase = blockIdx.y * 32;                // row in [0, B*T)

  const v8f vzero = {0.f, 0.f, 0.f, 0.f, 0.f, 0.f, 0.f, 0.f};
  v8f acc0 = vzero, acc1 = vzero;
  const __bf16* wrow  = wT + (size_t)n0 * C_;
  const __bf16* xrow  = xb + (size_t)rowBase * C_;

  prefetch_tile(&sx[0][0], xrow, C_);                 // chunk 0 -> buf 0
  const int NIT = C_ / 128;
  for (int it = 0; it < NIT; ++it) {
    const int p = it & 1;
    wait_async0();              // this wave's async LDS writes visible
    __syncthreads();            // everyone's writes visible, prev reads done
    if (it + 1 < NIT)
      prefetch_tile(&sx[p ^ 1][0], xrow + (it + 1) * 128, C_);
    const __bf16* sb = &sx[p][0];
#pragma unroll
    for (int kk = 0; kk < 4; ++kk) {
      v16bf b  = load_b_frag(wrow + it * 128 + kk * 32, C_);
      v16bf a0 = load_a_frag(sb + kk * 32, 128);
      v16bf a1 = load_a_frag(sb + 16 * 128 + kk * 32, 128);
      acc0 = wmma_bf16(a0, b, acc0);
      acc1 = wmma_bf16(a1, b, acc1);
    }
  }

  const int bb = rowBase >> 11;            // / T_
  const int t0 = rowBase & (T_ - 1);
  const int n  = n0 + col;
  qkv_epilogue(acc0, n, bb, t0,      hi, cosT, sinT, qO, kO, vTO);
  qkv_epilogue(acc1, n, bb, t0 + 16, hi, cosT, sinT, qO, kO, vTO);
}

// ---------------------------------------------------------------------------
// Kernel 3: flash attention.  1 wave = one 16-query tile of one head.
// 32-key steps: 8 WMMA (QK^T, K=128) + 8 WMMA (PV, Dh=128), online softmax
// with half-wave shfl_xor reductions; P transposed C->A layout through LDS.
// K/V fragments stream from global (L2-resident: ~1 MB per kv head).
// ---------------------------------------------------------------------------
__global__ void attn_kernel(const __bf16* __restrict__ qB,
                            const __bf16* __restrict__ kB,
                            const __bf16* __restrict__ vTB,
                            __bf16* __restrict__ ao,
                            const int* __restrict__ is_causal) {
  __shared__ alignas(32) __bf16 sp[4][16 * 32];   // per-wave P tile (16x32)

  const int wave = threadIdx.x >> 5;
  const int lane = threadIdx.x & 31;
  const int col  = lane & 15;
  const int hi   = lane >> 4;

  const int wg   = blockIdx.x * 4 + wave;   // global wave id
  const int tqt  = wg & (T_ / 16 - 1);      // query tile within head
  const int h    = (wg >> 7) & (H_ - 1);
  const int bb   = wg >> 12;                // / (128*32)
  const int kvh  = h >> 2;                  // G = 4
  const int tq0  = tqt * 16;

  const __bf16* qP  = qB  + ((size_t)(bb * H_ + h) * T_ + tq0) * DH_;
  const __bf16* kP  = kB  + (size_t)(bb * NKV_ + kvh) * T_ * DH_;
  const __bf16* vP  = vTB + (size_t)(bb * NKV_ + kvh) * DH_ * T_;
  __bf16*       spw = &sp[wave][0];

  // Q fragments held in registers for the whole key loop (Dh = 128 = 4x32)
  v16bf qf[4];
#pragma unroll
  for (int kk = 0; kk < 4; ++kk) qf[kk] = load_a_frag(qP + kk * 32, DH_);

  const v8f vzero = {0.f, 0.f, 0.f, 0.f, 0.f, 0.f, 0.f, 0.f};
  v8f o[8];
#pragma unroll
  for (int dt = 0; dt < 8; ++dt) o[dt] = vzero;
  float mrow[8], lrow[8];
#pragma unroll
  for (int r = 0; r < 8; ++r) { mrow[r] = -1e30f; lrow[r] = 0.f; }

  const bool causal = (is_causal[0] != 0);
  const int  kend   = causal ? (tq0 + 16) : T_;

  for (int kb = 0; kb < kend; kb += 32) {
    // ---- S = Q K^T (scaled already): two 16-key score tiles
    v8f s0 = vzero, s1 = vzero;
    const __bf16* kb0 = kP + (size_t)kb * DH_;
#pragma unroll
    for (int kk = 0; kk < 4; ++kk)
      s0 = wmma_bf16(qf[kk], load_b_frag(kb0 + kk * 32, DH_), s0);
#pragma unroll
    for (int kk = 0; kk < 4; ++kk)
      s1 = wmma_bf16(qf[kk], load_b_frag(kb0 + 16 * DH_ + kk * 32, DH_), s1);

    if (causal) {
#pragma unroll
      for (int r = 0; r < 8; ++r) {
        const int t = tq0 + r + 8 * hi;
        if (kb + col > t)      s0[r] = -1e30f;
        if (kb + 16 + col > t) s1[r] = -1e30f;
      }
    }

    // ---- online softmax (row reductions across each 16-lane half-wave)
    float alf[8];
#pragma unroll
    for (int r = 0; r < 8; ++r) {
      float mx = fmaxf(s0[r], s1[r]);
#pragma unroll
      for (int off = 1; off < 16; off <<= 1)
        mx = fmaxf(mx, __shfl_xor(mx, off, 32));
      const float mnew  = fmaxf(mrow[r], mx);
      const float alpha = __expf(mrow[r] - mnew);
      const float p0 = __expf(s0[r] - mnew);
      const float p1 = __expf(s1[r] - mnew);
      float rs = p0 + p1;
#pragma unroll
      for (int off = 1; off < 16; off <<= 1)
        rs += __shfl_xor(rs, off, 32);
      lrow[r] = lrow[r] * alpha + rs;
      mrow[r] = mnew;
      alf[r]  = alpha;
      s0[r] = p0;
      s1[r] = p1;
    }
#pragma unroll
    for (int dt = 0; dt < 8; ++dt) {
#pragma unroll
      for (int r = 0; r < 8; ++r) o[dt][r] *= alf[r];
    }

    // ---- P (C layout) -> LDS -> A fragment (16x32)
#pragma unroll
    for (int r = 0; r < 8; ++r) {
      const int row = r + 8 * hi;
      spw[row * 32 + col]      = (__bf16)s0[r];
      spw[row * 32 + 16 + col] = (__bf16)s1[r];
    }
    asm volatile("s_wait_dscnt 0" ::: "memory");   // wave-local LDS RAW fence
    v16bf pa = load_a_frag(spw, 32);

    // ---- O += P V : 8 d-tiles of 16, K = 32 keys (vT is [Dh][T])
#pragma unroll
    for (int dt = 0; dt < 8; ++dt)
      o[dt] = wmma_bf16(pa, load_b_frag(vP + (size_t)(dt * 16) * T_ + kb, T_),
                        o[dt]);
  }

  // ---- normalize and write out as bf16 rows of [B,T,C]
#pragma unroll
  for (int r = 0; r < 8; ++r) {
    const int t   = tq0 + r + 8 * hi;
    const float inv = 1.0f / lrow[r];
#pragma unroll
    for (int dt = 0; dt < 8; ++dt) {
      ao[((size_t)(bb * T_ + t)) * C_ + h * DH_ + dt * 16 + col] =
          (__bf16)(o[dt][r] * inv);
    }
  }
}

// ---------------------------------------------------------------------------
// Kernel 4: output projection  out = ao @ wo   (bf16 WMMA, f32 out).
// Same 32x128 double-buffered async tiling as kernel 2.
// ---------------------------------------------------------------------------
__global__ void out_gemm_kernel(const __bf16* __restrict__ ao,
                                const __bf16* __restrict__ woT,
                                float* __restrict__ out) {
  __shared__ alignas(32) __bf16 sx[2][32 * 128];

  const int wave    = threadIdx.x >> 5;
  const int lane    = threadIdx.x & 31;
  const int col     = lane & 15;
  const int hi      = lane >> 4;
  const int n0      = blockIdx.x * 128 + wave * 16;
  const int rowBase = blockIdx.y * 32;

  const v8f vzero = {0.f, 0.f, 0.f, 0.f, 0.f, 0.f, 0.f, 0.f};
  v8f acc0 = vzero, acc1 = vzero;
  const __bf16* wrow = woT + (size_t)n0 * C_;
  const __bf16* arow = ao + (size_t)rowBase * C_;

  prefetch_tile(&sx[0][0], arow, C_);
  const int NIT = C_ / 128;
  for (int it = 0; it < NIT; ++it) {
    const int p = it & 1;
    wait_async0();
    __syncthreads();
    if (it + 1 < NIT)
      prefetch_tile(&sx[p ^ 1][0], arow + (it + 1) * 128, C_);
    const __bf16* sb = &sx[p][0];
#pragma unroll
    for (int kk = 0; kk < 4; ++kk) {
      v16bf b  = load_b_frag(wrow + it * 128 + kk * 32, C_);
      v16bf a0 = load_a_frag(sb + kk * 32, 128);
      v16bf a1 = load_a_frag(sb + 16 * 128 + kk * 32, 128);
      acc0 = wmma_bf16(a0, b, acc0);
      acc1 = wmma_bf16(a1, b, acc1);
    }
  }

#pragma unroll
  for (int r = 0; r < 8; ++r) {
    out[(size_t)(rowBase + r + 8 * hi) * C_ + n0 + col]      = acc0[r];
    out[(size_t)(rowBase + 16 + r + 8 * hi) * C_ + n0 + col] = acc1[r];
  }
}

// ---------------------------------------------------------------------------
// Launch
// ---------------------------------------------------------------------------
extern "C" void kernel_launch(void* const* d_in, const int* in_sizes, int n_in,
                              void* d_out, int out_size, void* d_ws,
                              size_t ws_size, hipStream_t stream) {
  const float* x    = (const float*)d_in[0];
  const float* wq   = (const float*)d_in[1];
  const float* wk   = (const float*)d_in[2];
  const float* wv   = (const float*)d_in[3];
  const float* wo   = (const float*)d_in[4];
  const float* cosT = (const float*)d_in[5];
  const float* sinT = (const float*)d_in[6];
  const int* is_causal = (const int*)d_in[7];
  float* out = (float*)d_out;

  // Workspace layout (bytes).  `ao` (attention output, bf16 [B,T,C]) aliases
  // the xb region: xb is dead once the QKV GEMM finishes (stream-ordered).
  char* ws = (char*)d_ws;
  __bf16* xb    = (__bf16*)(ws);                    //  33,554,432  (also ao)
  __bf16* wqkvT = (__bf16*)(ws +  33554432ull);     //  50,331,648
  __bf16* woT   = (__bf16*)(ws +  83886080ull);     //  33,554,432
  __bf16* qb    = (__bf16*)(ws + 117440512ull);     //  33,554,432
  __bf16* kb    = (__bf16*)(ws + 150994944ull);     //   8,388,608
  __bf16* vT    = (__bf16*)(ws + 159383552ull);     //   8,388,608
  __bf16* ao    = xb;                               // total 160 MiB

  cvt_x_kernel<<<8192, 256, 0, stream>>>(x, xb, (size_t)B_ * T_ * C_);
  pack_w_kernel<<<16384, 256, 0, stream>>>(wq, wk, wv, wo, wqkvT, woT);
  qkv_rope_kernel<<<dim3(NQKV_ / 128, (B_ * T_) / 32), 256, 0, stream>>>(
      xb, wqkvT, cosT, sinT, qb, kb, vT);
  attn_kernel<<<(B_ * H_ * (T_ / 16)) / 4, 128, 0, stream>>>(qb, kb, vT, ao,
                                                             is_causal);
  out_gemm_kernel<<<dim3(C_ / 128, (B_ * T_) / 32), 256, 0, stream>>>(ao, woT,
                                                                      out);
}
